// TokenLinearFusion_31542239822249
// MI455X (gfx1250) — matrix operand order repaired and checked by reading, not compiled, not run on hardware
//
#include <hip/hip_runtime.h>
#include <hip/hip_bf16.h>
#include <math.h>

// ---------------------------------------------------------------------------
// Types for CDNA5 WMMA
// ---------------------------------------------------------------------------
typedef __bf16 v16bf __attribute__((ext_vector_type(16)));
typedef float  v8f   __attribute__((ext_vector_type(8)));
typedef unsigned int v8u __attribute__((ext_vector_type(8)));

#define T_TOK 256
#define ZDIM 16
#define HID 32
#define INF_DIM 1024
#define OUTF_DIM 1024
#define NROWS 32768

__device__ __forceinline__ float gelu_f(float v) {
    return 0.5f * v * (1.0f + erff(v * 0.7071067811865476f));
}

__device__ __forceinline__ unsigned short f2bf(float v) {
    __bf16 b = (__bf16)v;
    unsigned short u;
    __builtin_memcpy(&u, &b, 2);
    return u;
}

// ---------------------------------------------------------------------------
// Phase A: the whole token transformer (T=256, Z=16) in one block.
// Thread t owns token t. All f32 VALU; negligible cost.
// ---------------------------------------------------------------------------
__global__ __launch_bounds__(256) void tokens_kernel(
    const float* __restrict__ freq, const float* __restrict__ rnd,
    const float* __restrict__ w1, const float* __restrict__ b1,
    const float* __restrict__ w2, const float* __restrict__ b2,
    const float* __restrict__ wq, const float* __restrict__ wk,
    const float* __restrict__ wv, const float* __restrict__ bq,
    const float* __restrict__ bk, const float* __restrict__ bv,
    const float* __restrict__ wo, const float* __restrict__ bo,
    const float* __restrict__ ln1g, const float* __restrict__ ln1b,
    const float* __restrict__ ln2g, const float* __restrict__ ln2b,
    const float* __restrict__ mw1, const float* __restrict__ mb1,
    const float* __restrict__ mw2, const float* __restrict__ mb2,
    float* __restrict__ tokens_out)
{
    __shared__ float kbuf[T_TOK][ZDIM];
    __shared__ float vbuf[T_TOK][ZDIM];
    const int t = threadIdx.x;

    // --- token mixer ---
    float c[2 * ZDIM];
    for (int i = 0; i < ZDIM; ++i) { c[i] = freq[t * ZDIM + i]; c[ZDIM + i] = rnd[t * ZDIM + i]; }
    float h[HID];
    for (int j = 0; j < HID; ++j) {
        float s = b1[j];
        for (int i = 0; i < 2 * ZDIM; ++i) s += c[i] * w1[i * HID + j];
        h[j] = gelu_f(s);
    }
    float t0[ZDIM];
    for (int z = 0; z < ZDIM; ++z) {
        float s = b2[z];
        for (int j = 0; j < HID; ++j) s += h[j] * w2[j * ZDIM + z];
        t0[z] = s;
    }

    // --- q,k,v projections ---
    float q[ZDIM];
    for (int z = 0; z < ZDIM; ++z) {
        float sq = bq[z], sk = bk[z], sv = bv[z];
        for (int i = 0; i < ZDIM; ++i) {
            float ti = t0[i];
            sq += ti * wq[i * ZDIM + z];
            sk += ti * wk[i * ZDIM + z];
            sv += ti * wv[i * ZDIM + z];
        }
        q[z] = sq;
        kbuf[t][z] = sk; vbuf[t][z] = sv;
    }
    __syncthreads();

    // --- 2-head attention over 256 tokens (2-pass softmax) ---
    float ao[ZDIM];
    const float inv_sqrt_hd = 0.3535533905932738f; // 1/sqrt(8)
    for (int hd = 0; hd < 2; ++hd) {
        const int base = hd * 8;
        float mx = -INFINITY;
        for (int s = 0; s < T_TOK; ++s) {
            float d = 0.f;
            for (int e = 0; e < 8; ++e) d += q[base + e] * kbuf[s][base + e];
            mx = fmaxf(mx, d * inv_sqrt_hd);
        }
        float se = 0.f;
        float acc[8];
        for (int e = 0; e < 8; ++e) acc[e] = 0.f;
        for (int s = 0; s < T_TOK; ++s) {
            float d = 0.f;
            for (int e = 0; e < 8; ++e) d += q[base + e] * kbuf[s][base + e];
            float p = __expf(d * inv_sqrt_hd - mx);
            se += p;
            for (int e = 0; e < 8; ++e) acc[e] += p * vbuf[s][base + e];
        }
        float inv = 1.f / se;
        for (int e = 0; e < 8; ++e) ao[base + e] = acc[e] * inv;
    }

    // --- output proj + residual + LN1 ---
    float t1[ZDIM];
    {
        float r[ZDIM];
        for (int z = 0; z < ZDIM; ++z) {
            float s = bo[z];
            for (int i = 0; i < ZDIM; ++i) s += ao[i] * wo[i * ZDIM + z];
            r[z] = t0[z] + s;
        }
        float mu = 0.f;
        for (int z = 0; z < ZDIM; ++z) mu += r[z];
        mu *= (1.f / ZDIM);
        float var = 0.f;
        for (int z = 0; z < ZDIM; ++z) { float d = r[z] - mu; var += d * d; }
        var *= (1.f / ZDIM);
        float rs = rsqrtf(var + 1e-5f);
        for (int z = 0; z < ZDIM; ++z) t1[z] = (r[z] - mu) * rs * ln1g[z] + ln1b[z];
    }

    // --- MLP + residual + LN2 ---
    float t2[ZDIM];
    {
        float h2[4 * ZDIM];
        for (int j = 0; j < 4 * ZDIM; ++j) {
            float s = mb1[j];
            for (int z = 0; z < ZDIM; ++z) s += t1[z] * mw1[z * 4 * ZDIM + j];
            h2[j] = gelu_f(s);
        }
        float r[ZDIM];
        for (int z = 0; z < ZDIM; ++z) {
            float s = mb2[z];
            for (int j = 0; j < 4 * ZDIM; ++j) s += h2[j] * mw2[j * ZDIM + z];
            r[z] = t1[z] + s;
        }
        float mu = 0.f;
        for (int z = 0; z < ZDIM; ++z) mu += r[z];
        mu *= (1.f / ZDIM);
        float var = 0.f;
        for (int z = 0; z < ZDIM; ++z) { float d = r[z] - mu; var += d * d; }
        var *= (1.f / ZDIM);
        float rs = rsqrtf(var + 1e-5f);
        for (int z = 0; z < ZDIM; ++z) t2[z] = (r[z] - mu) * rs * ln2g[z] + ln2b[z];
    }

    for (int z = 0; z < ZDIM; ++z) tokens_out[t * ZDIM + z] = t2[z];
}

// ---------------------------------------------------------------------------
// Phase B: compute key_tokens / value_tokens (K=16 dot) and store them as
// bf16 in WMMA B-fragment layout:
//   flat = ((nt*KT + kt)*32 + lane)*16 + i
//   B element: N = nt*16 + (lane&15), K = kt*32 + 8*(lane>>4) + (i&7) + 16*(i>>3)
// key:   B is [K=feature(1024) x N=token(256)]  -> KT=32, NT=16, B[k][n]=key_tokens[n][k]
// value: B is [K=token(256)   x N=outf(1024)]   -> KT=8,  NT=64, B[k][n]=value_tokens[k][n]
// ---------------------------------------------------------------------------
__global__ __launch_bounds__(256) void pack_kernel(
    const float* __restrict__ tokens,
    const float* __restrict__ key_w,   // [16,1024] row-major
    const float* __restrict__ value_w, // [16,1024] row-major
    unsigned short* __restrict__ keyPk,
    unsigned short* __restrict__ valPk)
{
    const int tid = blockIdx.x * 256 + threadIdx.x; // 0 .. 2*262144-1
    const bool isVal = tid >= (T_TOK * INF_DIM);
    const int id = tid & (T_TOK * INF_DIM - 1);
    const int i = id & 15;
    const int lane = (id >> 4) & 31;

    int token, f;
    const float* w;
    unsigned short* outp;
    if (!isVal) {
        const int kt = (id >> 9) & 31;
        const int nt = id >> 14;            // 0..15
        token = nt * 16 + (lane & 15);      // N
        f     = kt * 32 + 8 * (lane >> 4) + (i & 7) + ((i >> 3) << 4); // K = feature
        w = key_w; outp = keyPk;
    } else {
        const int kt = (id >> 9) & 7;
        const int nt = id >> 12;            // 0..63
        token = kt * 32 + 8 * (lane >> 4) + (i & 7) + ((i >> 3) << 4); // K = token
        f     = nt * 16 + (lane & 15);      // N = out feature
        w = value_w; outp = valPk;
    }

    float s = 0.f;
    #pragma unroll
    for (int z = 0; z < ZDIM; ++z) s += tokens[token * ZDIM + z] * w[z * 1024 + f];
    outp[id] = f2bf(s);
}

// ---------------------------------------------------------------------------
// Phase C: fused  sim -> (gelu-norm, softmax) blend -> out GEMM.
// 128 threads = 4 waves, 16 rows of x per wave, 64 rows per block.
// ---------------------------------------------------------------------------
__global__ __launch_bounds__(128) void fuse_kernel(
    const float* __restrict__ x,
    const unsigned short* __restrict__ keyPk,
    const unsigned short* __restrict__ valPk,
    const float* __restrict__ alpha_logit,
    float* __restrict__ out)
{
    __shared__ unsigned short wbuf[4][16 * 256]; // per-wave 16x256 bf16 weights (32 KB)

    const int lane = threadIdx.x & 31;
    const int wave = threadIdx.x >> 5;
    const int row0 = blockIdx.x * 64 + wave * 16;
    const int m  = lane & 15;
    const int hi = lane >> 4;
    const int kb = hi * 8;

    const float* xrow = x + (size_t)(row0 + m) * INF_DIM;

    // ---- GEMM1: sim[16 x 256] = x_tile @ key_tokens^T, bf16 WMMA, f32 acc ----
    v8f acc[16];
    #pragma unroll
    for (int nt = 0; nt < 16; ++nt)
        #pragma unroll
        for (int r = 0; r < 8; ++r) acc[nt][r] = 0.f;

    for (int kt = 0; kt < 32; ++kt) {
        // Prefetch the x cacheline this lane needs 4 k-steps ahead
        // (one byte/lane -> global_prefetch_b8 pulls the full line).
        __builtin_prefetch(xrow + kt * 32 + kb + 128, 0, 3);

        // A fragment (16x32 bf16): lane holds row m, K = {kb..kb+7, 16+kb..16+kb+7}
        const float* p1 = xrow + kt * 32 + kb;
        const float4 ra = *(const float4*)(p1);
        const float4 rb = *(const float4*)(p1 + 4);
        const float4 rc = *(const float4*)(p1 + 16);
        const float4 rd = *(const float4*)(p1 + 20);
        v16bf a;
        a[0] = (__bf16)ra.x;  a[1] = (__bf16)ra.y;  a[2]  = (__bf16)ra.z;  a[3]  = (__bf16)ra.w;
        a[4] = (__bf16)rb.x;  a[5] = (__bf16)rb.y;  a[6]  = (__bf16)rb.z;  a[7]  = (__bf16)rb.w;
        a[8] = (__bf16)rc.x;  a[9] = (__bf16)rc.y;  a[10] = (__bf16)rc.z;  a[11] = (__bf16)rc.w;
        a[12] = (__bf16)rd.x; a[13] = (__bf16)rd.y; a[14] = (__bf16)rd.z;  a[15] = (__bf16)rd.w;

        #pragma unroll
        for (int nt = 0; nt < 16; ++nt) {
            const v8u braw = *(const v8u*)(keyPk + (((nt * 32 + kt) * 32 + lane) << 4));
            const v16bf b = __builtin_bit_cast(v16bf, braw);
            acc[nt] = __builtin_amdgcn_wmma_f32_16x16x32_bf16(
                false, a, false, b, (short)0, acc[nt], false, false);
        }
    }

    // ---- epilogue: per-row (over 256 cols) max / sumexp / sumsq, blend ----
    const float alpha = 1.f / (1.f + __expf(-alpha_logit[0]));
    const float one_m_alpha = 1.f - alpha;

    #pragma unroll
    for (int r = 0; r < 8; ++r) {
        // row M = r + 8*hi lives in this lane's half (16 lanes), cols N = nt*16 + m
        float mx = -INFINITY;
        #pragma unroll
        for (int nt = 0; nt < 16; ++nt) mx = fmaxf(mx, acc[nt][r]);
        #pragma unroll
        for (int off = 1; off < 16; off <<= 1) mx = fmaxf(mx, __shfl_xor(mx, off, 32));

        float g[16], e[16];
        float ss = 0.f, se = 0.f;
        #pragma unroll
        for (int nt = 0; nt < 16; ++nt) {
            const float a = acc[nt][r];
            const float gv = gelu_f(a * 16.0f);          // sim * sqrt(256)
            g[nt] = gv; ss += gv * gv;
            const float ev = __expf((a - mx) * (1.0f / 32.0f)); // softmax(sim/sqrt(1024))
            e[nt] = ev; se += ev;
        }
        #pragma unroll
        for (int off = 1; off < 16; off <<= 1) {
            ss += __shfl_xor(ss, off, 32);
            se += __shfl_xor(se, off, 32);
        }
        const float rg = alpha * rsqrtf(ss);
        const float rs = one_m_alpha / se;
        const int M = r + hi * 8;
        #pragma unroll
        for (int nt = 0; nt < 16; ++nt)
            wbuf[wave][M * 256 + nt * 16 + m] = f2bf(g[nt] * rg + e[nt] * rs);
    }
    __syncthreads();

    // ---- load weight A-fragments (16x32 each) from LDS ----
    const unsigned short* wb = wbuf[wave];
    v16bf aw[8];
    #pragma unroll
    for (int kt = 0; kt < 8; ++kt) {
        const int base = m * 256 + kt * 32 + kb;
        const uint4 u1 = *(const uint4*)(wb + base);
        const uint4 u2 = *(const uint4*)(wb + base + 16);
        v8u uu;
        uu[0] = u1.x; uu[1] = u1.y; uu[2] = u1.z; uu[3] = u1.w;
        uu[4] = u2.x; uu[5] = u2.y; uu[6] = u2.z; uu[7] = u2.w;
        aw[kt] = __builtin_bit_cast(v16bf, uu);
    }

    // ---- GEMM2: out[16 x 1024] = weights @ value_tokens ----
    for (int nt = 0; nt < 64; ++nt) {
        // warm WGP$ with the next n-tile's B fragments (L2-resident)
        __builtin_prefetch(valPk + ((((nt + 1) * 8) * 32 + lane) << 4), 0, 3);

        v8f o;
        #pragma unroll
        for (int r = 0; r < 8; ++r) o[r] = 0.f;
        #pragma unroll
        for (int kt = 0; kt < 8; ++kt) {
            const v8u braw = *(const v8u*)(valPk + (((nt * 8 + kt) * 32 + lane) << 4));
            const v16bf b = __builtin_bit_cast(v16bf, braw);
            o = __builtin_amdgcn_wmma_f32_16x16x32_bf16(
                false, aw[kt], false, b, (short)0, o, false, false);
        }
        #pragma unroll
        for (int r = 0; r < 8; ++r)
            out[(size_t)(row0 + hi * 8 + r) * OUTF_DIM + nt * 16 + m] = o[r];
    }
}

// ---------------------------------------------------------------------------
extern "C" void kernel_launch(void* const* d_in, const int* in_sizes, int n_in,
                              void* d_out, int out_size, void* d_ws, size_t ws_size,
                              hipStream_t stream) {
    const float* x       = (const float*)d_in[0];
    const float* freq    = (const float*)d_in[1];
    const float* rnd     = (const float*)d_in[2];
    const float* mix_w1  = (const float*)d_in[3];
    const float* mix_b1  = (const float*)d_in[4];
    const float* mix_w2  = (const float*)d_in[5];
    const float* mix_b2  = (const float*)d_in[6];
    const float* wq      = (const float*)d_in[7];
    const float* wk      = (const float*)d_in[8];
    const float* wv      = (const float*)d_in[9];
    const float* bq      = (const float*)d_in[10];
    const float* bk      = (const float*)d_in[11];
    const float* bv      = (const float*)d_in[12];
    const float* wo      = (const float*)d_in[13];
    const float* bo      = (const float*)d_in[14];
    const float* ln1g    = (const float*)d_in[15];
    const float* ln1b    = (const float*)d_in[16];
    const float* ln2g    = (const float*)d_in[17];
    const float* ln2b    = (const float*)d_in[18];
    const float* mlp_w1  = (const float*)d_in[19];
    const float* mlp_b1  = (const float*)d_in[20];
    const float* mlp_w2  = (const float*)d_in[21];
    const float* mlp_b2  = (const float*)d_in[22];
    const float* key_w   = (const float*)d_in[23];
    const float* value_w = (const float*)d_in[24];
    const float* alpha_l = (const float*)d_in[25];

    // workspace layout
    float* tokens = (float*)d_ws;                                          // 16 KB
    unsigned short* keyPk = (unsigned short*)((char*)d_ws + 16384);        // 512 KB
    unsigned short* valPk = (unsigned short*)((char*)d_ws + 16384 + 524288); // 512 KB

    tokens_kernel<<<1, 256, 0, stream>>>(
        freq, rnd, mix_w1, mix_b1, mix_w2, mix_b2,
        wq, wk, wv, bq, bk, bv, wo, bo,
        ln1g, ln1b, ln2g, ln2b,
        mlp_w1, mlp_b1, mlp_w2, mlp_b2, tokens);

    pack_kernel<<<(2 * T_TOK * INF_DIM) / 256, 256, 0, stream>>>(
        tokens, key_w, value_w, keyPk, valPk);

    fuse_kernel<<<NROWS / 64, 128, 0, stream>>>(
        x, keyPk, valPk, alpha_l, (float*)d_out);
}
